// block_47167330845113
// MI455X (gfx1250) — matrix operand order repaired and checked by reading, not compiled, not run on hardware
//
#include <hip/hip_runtime.h>
#include <hip/hip_bf16.h>
#include <math.h>

typedef __bf16 bf16_t;
typedef __attribute__((ext_vector_type(16))) __bf16 v16bf;
typedef __attribute__((ext_vector_type(8)))  __bf16 v8bf;
typedef __attribute__((ext_vector_type(8)))  float  v8f;
typedef __attribute__((ext_vector_type(4)))  int    i32x4;

#define WMMA_BF16(a, b, c) \
    __builtin_amdgcn_wmma_f32_16x16x32_bf16(false, (a), false, (b), (short)0, (c), false, false)

#if __has_builtin(__builtin_amdgcn_global_load_async_to_lds_b128)
#define HAS_ASYNC_LDS 1
#else
#define HAS_ASYNC_LDS 0
#endif

#if __has_builtin(__builtin_amdgcn_s_wait_asynccnt)
#define WAIT_ASYNC(n) __builtin_amdgcn_s_wait_asynccnt(n)
#else
#define WAIT_ASYNC(n) asm volatile("s_wait_asynccnt %0" ::"i"(n) : "memory")
#endif

// 16-byte global -> LDS copy: async DMA (ASYNCcnt) on CDNA5, sync fallback.
__device__ __forceinline__ void copy16_g2l(const void* g, void* l) {
#if HAS_ASYNC_LDS
    __builtin_amdgcn_global_load_async_to_lds_b128(
        (__attribute__((address_space(1))) i32x4*)g,
        (__attribute__((address_space(3))) i32x4*)l, 0, 0);
#else
    *(v8bf*)l = *(const v8bf*)g;
#endif
}

__device__ __forceinline__ v16bf ldfrag(const bf16_t* p0, const bf16_t* p1) {
    v8bf lo = *(const v8bf*)p0;
    v8bf hi = *(const v8bf*)p1;
    return __builtin_shufflevector(lo, hi, 0,1,2,3,4,5,6,7,8,9,10,11,12,13,14,15);
}

// ---------------------------------------------------------------------------
// Weight prep: fp32 W[K,N] -> bf16 Wt[N,K]
// ---------------------------------------------------------------------------
__global__ __launch_bounds__(256) void transpose_to_bf16(
    const float* __restrict__ w, bf16_t* __restrict__ wt, int K, int N) {
    __shared__ float tile[16][17];
    const int nb = blockIdx.x * 16, kb = blockIdx.y * 16;
    const int tx = threadIdx.x & 15, ty = threadIdx.x >> 4;
    tile[ty][tx] = w[(size_t)(kb + ty) * N + nb + tx];
    __syncthreads();
    wt[(size_t)(nb + ty) * K + kb + tx] = (bf16_t)tile[tx][ty];
}

// ---------------------------------------------------------------------------
// LayerNorm (fp32 in) -> bf16 out.  One 256-thread block per row, D=1024.
// ---------------------------------------------------------------------------
__global__ __launch_bounds__(256) void layernorm_bf16(
    const float* __restrict__ x, const float* __restrict__ w,
    const float* __restrict__ b, bf16_t* __restrict__ out) {
    const int D = 1024;
    const int row = blockIdx.x;
    const float* xr = x + (size_t)row * D;
    float v[4], s = 0.f, sq = 0.f;
#pragma unroll
    for (int i = 0; i < 4; ++i) {
        v[i] = xr[threadIdx.x + i * 256];
        s += v[i];
        sq += v[i] * v[i];
    }
    __shared__ float red0[256], red1[256];
    red0[threadIdx.x] = s; red1[threadIdx.x] = sq;
    __syncthreads();
    for (int off = 128; off > 0; off >>= 1) {
        if (threadIdx.x < off) {
            red0[threadIdx.x] += red0[threadIdx.x + off];
            red1[threadIdx.x] += red1[threadIdx.x + off];
        }
        __syncthreads();
    }
    const float mu = red0[0] * (1.0f / 1024.0f);
    const float var = red1[0] * (1.0f / 1024.0f) - mu * mu;
    const float rs = rsqrtf(var + 1e-5f);
#pragma unroll
    for (int i = 0; i < 4; ++i) {
        const int c = threadIdx.x + i * 256;
        out[(size_t)row * D + c] = (bf16_t)((v[i] - mu) * rs * w[c] + b[c]);
    }
}

// ---------------------------------------------------------------------------
// WMMA GEMM: C[M,N] = A[M,K](bf16) * Wt[N,K](bf16)^T + bias
//   MODE 0: store bf16 | MODE 1: GELU->bf16 | MODE 2: +resid, store fp32
// Tile 128x128xBK64, double-buffered LDS, async global->LDS staging.
// 8 waves, each wave = 64x32 (4x2 WMMA tiles).
// ---------------------------------------------------------------------------
template <int MODE>
__global__ __launch_bounds__(256) void gemm_bf16_wmma(
    const bf16_t* __restrict__ A, const bf16_t* __restrict__ Bt,
    const float* __restrict__ bias, const float* __restrict__ resid,
    void* __restrict__ outp, int M, int N, int K) {
    constexpr int BM = 128, BN = 128, BK = 64, LDP = BK + 8;
    __shared__ __align__(16) bf16_t As[2][BM][LDP];
    __shared__ __align__(16) bf16_t Bs[2][BN][LDP];

    const int tid = threadIdx.x;
    const int lane = tid & 31, wave = tid >> 5;
    const int wm = wave & 1, wn = wave >> 1;  // 2 (M) x 4 (N) waves
    const int m0 = blockIdx.y * BM, n0 = blockIdx.x * BN;
    const int r16 = lane & 15, hi = lane >> 4;

    v8f acc[4][2] = {};

    // stage loader: 4 chunks x (A,B) = 8 x 16B async copies per thread
    auto load_stage = [&](int bufi, int k0) {
#pragma unroll
        for (int c = 0; c < 4; ++c) {
            const int idx = c * 256 + tid;
            const int r = idx >> 3, ch = (idx & 7) * 8;
            copy16_g2l(&A[(size_t)(m0 + r) * K + k0 + ch], &As[bufi][r][ch]);
            copy16_g2l(&Bt[(size_t)(n0 + r) * K + k0 + ch], &Bs[bufi][r][ch]);
        }
    };

    load_stage(0, 0);
    int buf = 0;
    for (int k0 = 0; k0 < K; k0 += BK) {
        const bool pre = (k0 + BK < K);
        if (pre) load_stage(buf ^ 1, k0 + BK);  // overlap next stage DMA
#if HAS_ASYNC_LDS
        if (pre) WAIT_ASYNC(8);  // async loads retire in order: stage k0 done
        else     WAIT_ASYNC(0);
#endif
        __syncthreads();
#pragma unroll
        for (int kk = 0; kk < BK; kk += 32) {
            v16bf af[4], bfg[2];
#pragma unroll
            for (int mt = 0; mt < 4; ++mt) {
                const int r = wm * 64 + mt * 16 + r16;
                af[mt] = ldfrag(&As[buf][r][kk + hi * 8],
                                &As[buf][r][kk + hi * 8 + 16]);
            }
#pragma unroll
            for (int nt = 0; nt < 2; ++nt) {
                const int cI = wn * 32 + nt * 16 + r16;
                bfg[nt] = ldfrag(&Bs[buf][cI][kk + hi * 8],
                                 &Bs[buf][cI][kk + hi * 8 + 16]);
            }
#pragma unroll
            for (int mt = 0; mt < 4; ++mt)
#pragma unroll
                for (int nt = 0; nt < 2; ++nt)
                    acc[mt][nt] = WMMA_BF16(af[mt], bfg[nt], acc[mt][nt]);
        }
        __syncthreads();  // all waves done with buf before it is re-filled
        buf ^= 1;
    }

#pragma unroll
    for (int mt = 0; mt < 4; ++mt) {
#pragma unroll
        for (int nt = 0; nt < 2; ++nt) {
            const int col = n0 + wn * 32 + nt * 16 + r16;
            const float bc = bias[col];
#pragma unroll
            for (int j = 0; j < 8; ++j) {
                const int row = m0 + wm * 64 + mt * 16 + hi * 8 + j;
                float v = acc[mt][nt][j] + bc;
                if (MODE == 1)
                    v = 0.5f * v * (1.0f + erff(v * 0.70710678118654752f));
                if (MODE == 2)
                    ((float*)outp)[(size_t)row * N + col] =
                        v + resid[(size_t)row * N + col];
                else
                    ((bf16_t*)outp)[(size_t)row * N + col] = (bf16_t)v;
            }
        }
    }
}

// ---------------------------------------------------------------------------
// Flash attention (causal), bf16 qkv[B,L,3D] -> bf16 y[B,L,D].
// Grid: (L/128, B*H). 8 waves; wave owns 16 q-rows, online softmax.
// scale = 1/sqrt(D) = 1/32 (reference scales by embedding dim).
// ---------------------------------------------------------------------------
__global__ __launch_bounds__(256) void attention_wmma(
    const bf16_t* __restrict__ qkv, bf16_t* __restrict__ y) {
    const int D = 1024, H = 16, HD = 64, L = 2048;
    const int tid = threadIdx.x, lane = tid & 31, wave = tid >> 5;
    const int r16 = lane & 15, hi = lane >> 4;
    const int bh = blockIdx.y;
    const int b = bh / H, h = bh % H;
    const int qblk = blockIdx.x;
    const int qrow0 = qblk * 128 + wave * 16;
    const int rs3d = 3 * D;

    __shared__ __align__(16) bf16_t Ks[64][72];
    __shared__ __align__(16) bf16_t Vt[64][72];      // transposed: [hd][kc]
    __shared__ __align__(16) bf16_t Ps[8][16][72];   // per-wave P buffer

    const bf16_t* qbase = qkv + (size_t)(b * L) * rs3d + h * HD;
    const bf16_t* kbase = qbase + D;
    const bf16_t* vbase = qbase + 2 * D;

    v16bf qf[2];
#pragma unroll
    for (int s = 0; s < 2; ++s) {
        const bf16_t* qp = qbase + (size_t)(qrow0 + r16) * rs3d + s * 32 + hi * 8;
        qf[s] = ldfrag(qp, qp + 16);
    }

    v8f o[4] = {};
    float mrow[8], lrow[8];
#pragma unroll
    for (int j = 0; j < 8; ++j) { mrow[j] = -1e30f; lrow[j] = 0.f; }

    const int ktMax = 2 * qblk + 2;
    for (int kt = 0; kt < ktMax; ++kt) {
        __syncthreads();
#pragma unroll
        for (int c = 0; c < 2; ++c) {  // cooperative K/V tile load
            const int idx = c * 256 + tid;
            const int r = idx >> 3, ch = (idx & 7) * 8;
            copy16_g2l(&kbase[(size_t)(kt * 64 + r) * rs3d + ch], &Ks[r][ch]);
            v8bf vv = *(const v8bf*)&vbase[(size_t)(kt * 64 + r) * rs3d + ch];
#pragma unroll
            for (int e = 0; e < 8; ++e) Vt[ch + e][r] = vv[e];  // transpose V
        }
#if HAS_ASYNC_LDS
        WAIT_ASYNC(0);
#endif
        __syncthreads();
        if (kt * 64 > qrow0 + 15) continue;  // fully above diagonal

        // S = q @ k^T  (16 x 64 per wave)
        v8f s[4] = {};
#pragma unroll
        for (int nt = 0; nt < 4; ++nt) {
#pragma unroll
            for (int ks = 0; ks < 2; ++ks) {
                const int cI = nt * 16 + r16;
                v16bf kb = ldfrag(&Ks[cI][ks * 32 + hi * 8],
                                  &Ks[cI][ks * 32 + hi * 8 + 16]);
                s[nt] = WMMA_BF16(qf[ks], kb, s[nt]);
            }
        }

        // scale + causal mask + online softmax (row = hi*8+j, col = lanes)
        float newm[8];
#pragma unroll
        for (int j = 0; j < 8; ++j) {
            const int qr = qrow0 + hi * 8 + j;
            float mx = -1e30f;
#pragma unroll
            for (int nt = 0; nt < 4; ++nt) {
                const int key = kt * 64 + nt * 16 + r16;
                float sv = s[nt][j] * 0.03125f;
                sv = (key <= qr) ? sv : -1e30f;
                s[nt][j] = sv;
                mx = fmaxf(mx, sv);
            }
#pragma unroll
            for (int m = 1; m < 16; m <<= 1) mx = fmaxf(mx, __shfl_xor(mx, m, 32));
            newm[j] = fmaxf(mrow[j], mx);
        }
#pragma unroll
        for (int j = 0; j < 8; ++j) {
            const float alpha = expf(mrow[j] - newm[j]);
            float rsum = 0.f;
#pragma unroll
            for (int nt = 0; nt < 4; ++nt) {
                const float p = expf(s[nt][j] - newm[j]);
                rsum += p;
                Ps[wave][hi * 8 + j][nt * 16 + r16] = (bf16_t)p;
            }
#pragma unroll
            for (int m = 1; m < 16; m <<= 1) rsum += __shfl_xor(rsum, m, 32);
            lrow[j] = lrow[j] * alpha + rsum;
            mrow[j] = newm[j];
#pragma unroll
            for (int t = 0; t < 4; ++t) o[t][j] *= alpha;
        }

        // O += P @ V  (P via per-wave LDS to reshape C-layout -> A-layout)
#pragma unroll
        for (int ks = 0; ks < 2; ++ks) {
            v16bf pf = ldfrag(&Ps[wave][r16][ks * 32 + hi * 8],
                              &Ps[wave][r16][ks * 32 + hi * 8 + 16]);
#pragma unroll
            for (int t = 0; t < 4; ++t) {
                const int cI = t * 16 + r16;
                v16bf vb = ldfrag(&Vt[cI][ks * 32 + hi * 8],
                                  &Vt[cI][ks * 32 + hi * 8 + 16]);
                o[t] = WMMA_BF16(pf, vb, o[t]);
            }
        }
    }

#pragma unroll
    for (int t = 0; t < 4; ++t) {
#pragma unroll
        for (int j = 0; j < 8; ++j) {
            const int qr = qrow0 + hi * 8 + j;
            y[(size_t)(b * L + qr) * D + h * HD + t * 16 + r16] =
                (bf16_t)(o[t][j] / lrow[j]);
        }
    }
}

// ---------------------------------------------------------------------------
extern "C" void kernel_launch(void* const* d_in, const int* in_sizes, int n_in,
                              void* d_out, int out_size, void* d_ws,
                              size_t ws_size, hipStream_t stream) {
    (void)in_sizes; (void)n_in; (void)out_size; (void)ws_size;
    const int Bsz = 2, L = 2048;
    const int M = Bsz * L;  // 4096

    const float* x      = (const float*)d_in[0];
    const float* ln1w   = (const float*)d_in[1];
    const float* ln1b   = (const float*)d_in[2];
    const float* ln2w   = (const float*)d_in[3];
    const float* ln2b   = (const float*)d_in[4];
    const float* attn_w = (const float*)d_in[5];
    const float* attn_b = (const float*)d_in[6];
    const float* proj_w = (const float*)d_in[7];
    const float* proj_b = (const float*)d_in[8];
    const float* fc1_w  = (const float*)d_in[9];
    const float* fc1_b  = (const float*)d_in[10];
    const float* fc2_w  = (const float*)d_in[11];
    const float* fc2_b  = (const float*)d_in[12];
    float* out = (float*)d_out;

    char* wsb = (char*)d_ws;
    size_t off = 0;
    auto alloc = [&](size_t bytes) -> void* {
        void* p = wsb + off;
        off = (off + bytes + 255) & ~(size_t)255;
        return p;
    };
    bf16_t* wqkvt = (bf16_t*)alloc((size_t)3072 * 1024 * 2);
    bf16_t* wprjt = (bf16_t*)alloc((size_t)1024 * 1024 * 2);
    bf16_t* wfc1t = (bf16_t*)alloc((size_t)4096 * 1024 * 2);
    bf16_t* wfc2t = (bf16_t*)alloc((size_t)1024 * 4096 * 2);
    bf16_t* hbf   = (bf16_t*)alloc((size_t)M * 1024 * 2);
    bf16_t* qkvb  = (bf16_t*)alloc((size_t)M * 3072 * 2);
    bf16_t* ybf   = (bf16_t*)alloc((size_t)M * 1024 * 2);
    bf16_t* h2bf  = (bf16_t*)alloc((size_t)M * 1024 * 2);
    bf16_t* a1bf  = (bf16_t*)alloc((size_t)M * 4096 * 2);

    transpose_to_bf16<<<dim3(192, 64), 256, 0, stream>>>(attn_w, wqkvt, 1024, 3072);
    transpose_to_bf16<<<dim3(64, 64), 256, 0, stream>>>(proj_w, wprjt, 1024, 1024);
    transpose_to_bf16<<<dim3(256, 64), 256, 0, stream>>>(fc1_w, wfc1t, 1024, 4096);
    transpose_to_bf16<<<dim3(64, 256), 256, 0, stream>>>(fc2_w, wfc2t, 4096, 1024);

    layernorm_bf16<<<M, 256, 0, stream>>>(x, ln1w, ln1b, hbf);

    gemm_bf16_wmma<0><<<dim3(24, 32), 256, 0, stream>>>(
        hbf, wqkvt, attn_b, nullptr, qkvb, M, 3072, 1024);

    attention_wmma<<<dim3(16, 32), 256, 0, stream>>>(qkvb, ybf);

    gemm_bf16_wmma<2><<<dim3(8, 32), 256, 0, stream>>>(
        ybf, wprjt, proj_b, x, out, M, 1024, 1024);

    layernorm_bf16<<<M, 256, 0, stream>>>(out, ln2w, ln2b, h2bf);

    gemm_bf16_wmma<1><<<dim3(32, 32), 256, 0, stream>>>(
        h2bf, wfc1t, fc1_b, nullptr, a1bf, M, 4096, 1024);

    gemm_bf16_wmma<2><<<dim3(8, 32), 256, 0, stream>>>(
        a1bf, wfc2t, fc2_b, out, out, M, 1024, 4096);
}
